// Attention4D_12987981103351
// MI455X (gfx1250) — matrix operand order repaired and checked by reading, not compiled, not run on hardware
//
#include <hip/hip_runtime.h>

typedef __bf16 bf16_t;
typedef __attribute__((ext_vector_type(16))) __bf16 v16bf;
typedef __attribute__((ext_vector_type(8)))  __bf16 v8bf;
typedef __attribute__((ext_vector_type(8)))  float  v8f;

#define RES_ 14
#define N_   196
#define NH_  8
#define KD_  32
#define D_   128
#define DH_  1024
#define DIM_ 384
#define B_   128
#define NP_  208   // tokens padded to 13*16
#define NP2_ 224   // tokens padded to 7*32 (K dim for attn@V)
#define SCALE_ 0.17677669529663687f  // 32^-0.5

// LDS layout for the GEMM kernels (single __shared__ object => allocated at LDS offset 0)
#define LDS_X0 0u
#define LDS_X1 13312u
#define LDS_W0 26624u
#define LDS_W1 34816u
#define LDS_TOTAL 43008

// ---------------- WMMA fragment helpers (wave32, 16-bit layouts per ISA 7.12.2) --------
// A (16x32, MxK): lanes 0-15 row M=lane, K={0..7,16..23}; lanes 16-31 row M=lane-16, K={8..15,24..31}
__device__ __forceinline__ v16bf load_frag_a(const bf16_t* base, int rowStride, int k0) {
  const int lane = threadIdx.x & 31;
  const int half = lane >> 4;
  const int m    = lane & 15;
  const bf16_t* p = base + (size_t)m * rowStride + k0 + half * 8;
  v8bf lo = *(const v8bf*)(p);
  v8bf hi = *(const v8bf*)(p + 16);
  v16bf a;
#pragma unroll
  for (int t = 0; t < 8; ++t) { a[t] = lo[t]; a[t + 8] = hi[t]; }
  return a;
}

// B (32x16, KxN), memory tile layout [n][K...] with stride colStride.
// lanes 0-15: col N=lane, K={0..7,16..23} ; lanes 16-31: col N=lane-16, K={8..15,24..31}
__device__ __forceinline__ v16bf load_frag_b(const bf16_t* base, int colStride, int k0) {
  const int lane = threadIdx.x & 31;
  const int half = lane >> 4;
  const int n    = lane & 15;
  const bf16_t* p = base + (size_t)n * colStride + k0 + half * 8;
  v8bf lo = *(const v8bf*)(p);
  v8bf hi = *(const v8bf*)(p + 16);
  v16bf b;
#pragma unroll
  for (int t = 0; t < 8; ++t) { b[t] = lo[t]; b[t + 8] = hi[t]; }
  return b;
}

__device__ __forceinline__ v8f zero_v8f() {
  v8f z;
#pragma unroll
  for (int t = 0; t < 8; ++t) z[t] = 0.f;
  return z;
}

__device__ __forceinline__ v8f wmma_bf16(v16bf a, v16bf b, v8f c) {
  return __builtin_amdgcn_wmma_f32_16x16x32_bf16(false, a, false, b, (short)0, c, false, false);
}

// Make an LDS object's address escape into asm so that subsequent asm "memory"
// clobbers (the async copies) are known to alias it; otherwise the compiler
// folds loads from an LDS buffer that has no visible stores to undef.
__device__ __forceinline__ void lds_escape(void* p) {
  asm volatile("" : "+v"(p) :: "memory");
}

// ---- async global->LDS panel copy (gfx1250 GLOBAL_LOAD_ASYNC_TO_LDS_B128, ASYNCcnt) ----
// Copies nRows rows of 64 bytes (32 bf16) from src (row stride rowStrideElems) into an
// LDS tile laid out [row][32] at byte offset ldsOff. 16-byte chunks, contiguity-matched.
__device__ __forceinline__ void async_copy_panel(const bf16_t* src, int rowStrideElems,
                                                 int nRows, unsigned ldsOff) {
  const int tid = threadIdx.x;
  const int total = nRows * 4;
  unsigned long long base = (unsigned long long)(size_t)src;
  for (int idx = tid; idx < total; idx += 256) {
    int r = idx >> 2, t = idx & 3;
    unsigned lo = ldsOff + (unsigned)(r * 64 + t * 16);
    unsigned go = (unsigned)((r * rowStrideElems + t * 8) * 2);
    asm volatile("global_load_async_to_lds_b128 %0, %1, %2"
                 :: "v"(lo), "v"(go), "s"(base) : "memory");
  }
}

__device__ __forceinline__ void wait_async_all() {
  asm volatile("s_wait_asynccnt 0x0" ::: "memory");
}

// ---------------- kernel 0: fold BN affine into weights, convert to bf16 ----------------
__global__ __launch_bounds__(256) void prep_kernel(
    const float* __restrict__ qw, const float* __restrict__ qb2, const float* __restrict__ q_s, const float* __restrict__ q_t,
    const float* __restrict__ kw, const float* __restrict__ kb2, const float* __restrict__ k_s, const float* __restrict__ k_t,
    const float* __restrict__ vw, const float* __restrict__ vb2, const float* __restrict__ v_s, const float* __restrict__ v_t,
    const float* __restrict__ lvw, const float* __restrict__ lvb, const float* __restrict__ lv_s, const float* __restrict__ lv_t,
    const float* __restrict__ pw, const float* __restrict__ pb, const float* __restrict__ p_s, const float* __restrict__ p_t,
    bf16_t* __restrict__ wqkv, float* __restrict__ bqkv, bf16_t* __restrict__ pwb,
    float* __restrict__ pbf, float* __restrict__ lvwf, float* __restrict__ lvbf)
{
  const int R0 = 1536 * DIM_;       // qkv weights
  const int R1 = R0 + DIM_ * DH_;   // pw
  const int R2 = R1 + DH_ * 9;      // lvw
  const int R3 = R2 + 1536;         // bqkv
  const int R4 = R3 + DIM_;         // pb
  const int R5 = R4 + DH_;          // lvb
  int idx = blockIdx.x * 256 + threadIdx.x;
  if (idx < R0) {
    int m = idx / DIM_, c = idx % DIM_;
    float w, s;
    if (m < 256)      { w = qw[idx];              s = q_s[m]; }
    else if (m < 512) { w = kw[(m-256)*DIM_ + c]; s = k_s[m-256]; }
    else              { w = vw[(m-512)*DIM_ + c]; s = v_s[m-512]; }
    wqkv[idx] = (bf16_t)(w * s);
  } else if (idx < R1) {
    int i = idx - R0;
    pwb[i] = (bf16_t)(pw[i] * p_s[i / DH_]);
  } else if (idx < R2) {
    int i = idx - R1;
    lvwf[i] = lvw[i] * lv_s[i / 9];
  } else if (idx < R3) {
    int m = idx - R2;
    float bv, s, t;
    if (m < 256)      { bv = qb2[m];     s = q_s[m];     t = q_t[m]; }
    else if (m < 512) { bv = kb2[m-256]; s = k_s[m-256]; t = k_t[m-256]; }
    else              { bv = vb2[m-512]; s = v_s[m-512]; t = v_t[m-512]; }
    bqkv[m] = bv * s + t;
  } else if (idx < R4) {
    int o = idx - R3;
    pbf[o] = pb[o] * p_s[o] + p_t[o];
  } else if (idx < R5) {
    int c = idx - R4;
    lvbf[c] = lvb[c] * lv_s[c] + lv_t[c];
  }
}

// ---------------- kernel 0b: transpose x to token-major bf16  xbf[b][208][384] ----------
__global__ __launch_bounds__(256) void xpose_kernel(const float* __restrict__ x,
                                                    bf16_t* __restrict__ xbf)
{
  __shared__ bf16_t t[32 * NP_];
  const int b = blockIdx.x, c0 = blockIdx.y * 32;
  const int tid = threadIdx.x;
  for (int idx = tid; idx < 32 * NP_; idx += 256) {
    int c = idx / NP_, n = idx % NP_;
    float v = (n < N_) ? x[((size_t)b * DIM_ + c0 + c) * N_ + n] : 0.f;
    t[c * NP_ + n] = (bf16_t)v;
  }
  __syncthreads();
  for (int idx = tid; idx < 32 * NP_; idx += 256) {
    int n = idx >> 5, c = idx & 31;
    xbf[((size_t)b * NP_ + n) * DIM_ + c0 + c] = t[c * NP_ + n];
  }
}

// ---------------- kernel 1: QKV projection GEMM  Y = Wqkv(1536x384) @ X_b(384x208) ------
// grid (B, 12 mBlocks of 128 rows); 8 waves, wave w owns 16-row mTile, all 13 nTiles.
// Double-buffered async global->LDS staging for both panels.
__global__ __launch_bounds__(256) void qkv_gemm(
    const bf16_t* __restrict__ xbf, const bf16_t* __restrict__ wqkv, const float* __restrict__ bqkv,
    bf16_t* __restrict__ qbuf, bf16_t* __restrict__ kbuf, bf16_t* __restrict__ vbuf)
{
  __shared__ __align__(16) char ldsbuf[LDS_TOTAL];
  lds_escape(ldsbuf);

  const int b = blockIdx.x, mBlock = blockIdx.y;
  const int tid = threadIdx.x, wave = tid >> 5, lane = tid & 31;
  const int half = lane >> 4, ln = lane & 15;

  const bf16_t* xsrc = xbf  + (size_t)b * NP_ * DIM_;
  const bf16_t* wsrc = wqkv + (size_t)(mBlock * 128) * DIM_;

  v8f acc[13];
#pragma unroll
  for (int t = 0; t < 13; ++t) acc[t] = zero_v8f();

  async_copy_panel(xsrc, DIM_, NP_, LDS_X0);
  async_copy_panel(wsrc, DIM_, 128, LDS_W0);
  wait_async_all();
  __syncthreads();

  for (int kc = 0; kc < 12; ++kc) {
    const int cur = kc & 1;
    if (kc + 1 < 12) {
      async_copy_panel(xsrc + (kc + 1) * 32, DIM_, NP_, cur ? LDS_X0 : LDS_X1);
      async_copy_panel(wsrc + (kc + 1) * 32, DIM_, 128, cur ? LDS_W0 : LDS_W1);
    }
    bf16_t* ldsX = (bf16_t*)(ldsbuf + (cur ? LDS_X1 : LDS_X0));
    bf16_t* ldsW = (bf16_t*)(ldsbuf + (cur ? LDS_W1 : LDS_W0));
    v16bf a = load_frag_a(ldsW + wave * 16 * 32, 32, 0);
#pragma unroll
    for (int nt = 0; nt < 13; ++nt) {
      v16bf bf = load_frag_b(ldsX + nt * 16 * 32, 32, 0);
      acc[nt] = wmma_bf16(a, bf, acc[nt]);
    }
    wait_async_all();
    __syncthreads();
  }

  const int mBase = mBlock * 128 + wave * 16;
#pragma unroll
  for (int nt = 0; nt < 13; ++nt) {
#pragma unroll
    for (int r = 0; r < 8; ++r) {
      int m = mBase + half * 8 + r;
      int n = nt * 16 + ln;
      bf16_t bv = (bf16_t)(acc[nt][r] + bqkv[m]);
      if (m < 256) {
        int h = m >> 5, kd = m & 31;
        qbuf[(((size_t)b * NH_ + h) * NP_ + n) * KD_ + kd] = bv;
      } else if (m < 512) {
        int mm = m - 256, h = mm >> 5, kd = mm & 31;
        kbuf[(((size_t)b * NH_ + h) * NP_ + n) * KD_ + kd] = bv;
      } else {
        vbuf[((size_t)b * DH_ + (m - 512)) * NP2_ + n] = bv;
      }
    }
  }
  // zero the v padding columns 208..223 so attn@V K-padding is clean
  if (mBase >= 512) {
#pragma unroll
    for (int r = 0; r < 8; ++r) {
      int c = mBase + half * 8 + r - 512;
      vbuf[((size_t)b * DH_ + c) * NP2_ + 208 + ln] = (bf16_t)0.f;
    }
  }
}

// ---------------- kernel 2: depthwise 3x3 conv on v (folded BN) -------------------------
__global__ __launch_bounds__(256) void dwconv_kernel(
    const bf16_t* __restrict__ vbuf, const float* __restrict__ lvwf,
    const float* __restrict__ lvbf, bf16_t* __restrict__ vlb)
{
  size_t idx = (size_t)blockIdx.x * 256 + threadIdx.x;
  if (idx >= (size_t)B_ * DH_ * NP_) return;
  int n = (int)(idx % NP_);
  size_t bc = idx / NP_;
  int c = (int)(bc % DH_);
  int b = (int)(bc / DH_);
  float r = 0.f;
  if (n < N_) {
    int y = n / RES_, x0 = n % RES_;
    r = lvbf[c];
    const bf16_t* vsrc = vbuf + ((size_t)b * DH_ + c) * NP2_;
    const float* wk = lvwf + c * 9;
#pragma unroll
    for (int dy = -1; dy <= 1; ++dy)
#pragma unroll
      for (int dx = -1; dx <= 1; ++dx) {
        int yy = y + dy, xx = x0 + dx;
        if (yy >= 0 && yy < RES_ && xx >= 0 && xx < RES_)
          r += (float)vsrc[yy * RES_ + xx] * wk[(dy + 1) * 3 + (dx + 1)];
      }
  }
  vlb[idx] = (bf16_t)r;
}

// ---------------- kernel 3: fused attention core ---------------------------------------
// grid (B, 13 iTiles); 8 waves = 8 heads. LDS: S[8][16][224] f32 + At[8][16][224] bf16.
__global__ __launch_bounds__(256) void attn_kernel(
    const bf16_t* __restrict__ qbuf, const bf16_t* __restrict__ kbuf,
    const bf16_t* __restrict__ vbuf, const bf16_t* __restrict__ vlb,
    const float* __restrict__ ab, const float* __restrict__ th1w, const float* __restrict__ th1b,
    const float* __restrict__ th2w, const float* __restrict__ th2b,
    bf16_t* __restrict__ obuf)
{
  extern __shared__ char smem[];
  float*  S  = (float*)smem;                          // 8*16*224*4 = 114688 B
  bf16_t* At = (bf16_t*)(smem + 8 * 16 * NP2_ * 4);   // 8*16*224*2 = 57344 B
  const int b = blockIdx.x, i0 = blockIdx.y * 16;
  const int tid = threadIdx.x, wave = tid >> 5, lane = tid & 31;
  const int half = lane >> 4, ln = lane & 15;
  const int h = wave;

  // --- S = (q k^T)*scale + rel-pos bias ---
  const bf16_t* qh = qbuf + ((size_t)b * NH_ + h) * NP_ * KD_;
  const bf16_t* kh = kbuf + ((size_t)b * NH_ + h) * NP_ * KD_;
  v16bf aq = load_frag_a(qh + (size_t)i0 * KD_, KD_, 0);
#pragma unroll
  for (int jt = 0; jt < 13; ++jt) {
    v16bf bk = load_frag_b(kh + (size_t)jt * 16 * KD_, KD_, 0);
    v8f s = wmma_bf16(aq, bk, zero_v8f());
#pragma unroll
    for (int r = 0; r < 8; ++r) {
      int i = half * 8 + r;
      int itok = i0 + i;
      int j = jt * 16 + ln;
      float val;
      if (j < N_) {
        int ii = (itok < N_) ? itok : 0;
        int ai = ii / RES_, aj = ii % RES_;
        int bi = j / RES_,  bj = j % RES_;
        int dy = ai - bi; if (dy < 0) dy = -dy;
        int dx = aj - bj; if (dx < 0) dx = -dx;
        val = s[r] * SCALE_ + ab[h * N_ + dy * RES_ + dx];
      } else {
        val = -1e30f;
      }
      S[(h * 16 + i) * NP2_ + j] = val;
    }
  }
  __syncthreads();

  // --- talking head 1 (mix across heads; each thread owns distinct (i,j)) ---
  for (int idx = tid; idx < 16 * NP_; idx += 256) {
    int i = idx / NP_, j = idx % NP_;
    float vh[8], vg[8];
    if (j < N_) {
#pragma unroll
      for (int hh = 0; hh < 8; ++hh) vh[hh] = S[(hh * 16 + i) * NP2_ + j];
#pragma unroll
      for (int g = 0; g < 8; ++g) {
        float a0 = th1b[g];
#pragma unroll
        for (int hh = 0; hh < 8; ++hh) a0 += vh[hh] * th1w[g * 8 + hh];
        vg[g] = a0;
      }
    } else {
#pragma unroll
      for (int g = 0; g < 8; ++g) vg[g] = -1e30f;
    }
#pragma unroll
    for (int g = 0; g < 8; ++g) S[(g * 16 + i) * NP2_ + j] = vg[g];
  }
  __syncthreads();

  // --- softmax over j (one row per thread) ---
  if (tid < 128) {
    int g = tid >> 4, i = tid & 15;
    float* row = S + (g * 16 + i) * NP2_;
    float m = -1e30f;
    for (int j = 0; j < N_; ++j) m = fmaxf(m, row[j]);
    float sum = 0.f;
    for (int j = 0; j < N_; ++j) { float e = __expf(row[j] - m); row[j] = e; sum += e; }
    float inv = 1.f / sum;
    for (int j = 0; j < N_; ++j) row[j] *= inv;
  }
  __syncthreads();

  // --- talking head 2 + convert to bf16 attention panel (zero K-padding) ---
  for (int idx = tid; idx < 16 * NP2_; idx += 256) {
    int i = idx / NP2_, j = idx % NP2_;
    if (j < N_) {
      float vh[8];
#pragma unroll
      for (int hh = 0; hh < 8; ++hh) vh[hh] = S[(hh * 16 + i) * NP2_ + j];
#pragma unroll
      for (int g = 0; g < 8; ++g) {
        float a0 = th2b[g];
#pragma unroll
        for (int hh = 0; hh < 8; ++hh) a0 += vh[hh] * th2w[g * 8 + hh];
        At[(g * 16 + i) * NP2_ + j] = (bf16_t)a0;
      }
    } else {
#pragma unroll
      for (int g = 0; g < 8; ++g) At[(g * 16 + i) * NP2_ + j] = (bf16_t)0.f;
    }
  }
  __syncthreads();

  // --- O = attn @ V, + vl, ReLU -> bf16 (token-major obuf[b][208][1024] for out-proj) ---
  const bf16_t* vh   = vbuf + ((size_t)b * DH_ + h * D_) * NP2_;   // [128 d][224 j]
  const bf16_t* vlh  = vlb  + ((size_t)b * DH_ + h * D_) * NP_;
  bf16_t*       ob   = obuf + (size_t)b * NP_ * DH_;
  const bf16_t* apan = At + (size_t)(h * 16) * NP2_;
#pragma unroll
  for (int dt = 0; dt < 8; ++dt) {
    v8f acc = zero_v8f();
#pragma unroll
    for (int jc = 0; jc < 7; ++jc) {
      v16bf afr = load_frag_a(apan + jc * 32, NP2_, 0);
      v16bf bfr = load_frag_b(vh + (size_t)dt * 16 * NP2_, NP2_, jc * 32);
      acc = wmma_bf16(afr, bfr, acc);
    }
#pragma unroll
    for (int r = 0; r < 8; ++r) {
      int i = half * 8 + r;
      int tok = i0 + i;
      int d = dt * 16 + ln;
      float val = acc[r] + (float)vlh[(size_t)d * NP_ + tok];
      val = fmaxf(val, 0.f);
      ob[(size_t)tok * DH_ + h * D_ + d] = (bf16_t)val;
    }
  }
}

// ---------------- kernel 4: output projection  out = pw(384x1024) @ orelu(1024x208) -----
// grid (B, 3 mBlocks of 128 rows); double-buffered async staging for both panels.
__global__ __launch_bounds__(256) void proj_kernel(
    const bf16_t* __restrict__ obuf, const bf16_t* __restrict__ pwb,
    const float* __restrict__ pbf, float* __restrict__ out)
{
  __shared__ __align__(16) char ldsbuf[LDS_TOTAL];
  lds_escape(ldsbuf);

  const int b = blockIdx.x, mBlock = blockIdx.y;
  const int tid = threadIdx.x, wave = tid >> 5, lane = tid & 31;
  const int half = lane >> 4, ln = lane & 15;

  const bf16_t* xsrc = obuf + (size_t)b * NP_ * DH_;
  const bf16_t* wsrc = pwb  + (size_t)(mBlock * 128) * DH_;

  v8f acc[13];
#pragma unroll
  for (int t = 0; t < 13; ++t) acc[t] = zero_v8f();

  async_copy_panel(xsrc, DH_, NP_, LDS_X0);
  async_copy_panel(wsrc, DH_, 128, LDS_W0);
  wait_async_all();
  __syncthreads();

  for (int kc = 0; kc < 32; ++kc) {
    const int cur = kc & 1;
    if (kc + 1 < 32) {
      async_copy_panel(xsrc + (kc + 1) * 32, DH_, NP_, cur ? LDS_X0 : LDS_X1);
      async_copy_panel(wsrc + (kc + 1) * 32, DH_, 128, cur ? LDS_W0 : LDS_W1);
    }
    bf16_t* ldsX = (bf16_t*)(ldsbuf + (cur ? LDS_X1 : LDS_X0));
    bf16_t* ldsW = (bf16_t*)(ldsbuf + (cur ? LDS_W1 : LDS_W0));
    v16bf a = load_frag_a(ldsW + wave * 16 * 32, 32, 0);
#pragma unroll
    for (int nt = 0; nt < 13; ++nt) {
      v16bf bf = load_frag_b(ldsX + nt * 16 * 32, 32, 0);
      acc[nt] = wmma_bf16(a, bf, acc[nt]);
    }
    wait_async_all();
    __syncthreads();
  }

  const int mBase = mBlock * 128 + wave * 16;
#pragma unroll
  for (int nt = 0; nt < 13; ++nt) {
#pragma unroll
    for (int r = 0; r < 8; ++r) {
      int m = mBase + half * 8 + r;
      int n = nt * 16 + ln;
      if (n < N_)
        out[((size_t)b * DIM_ + m) * N_ + n] = acc[nt][r] + pbf[m];
    }
  }
}

// ---------------- launch ---------------------------------------------------------------
extern "C" void kernel_launch(void* const* d_in, const int* in_sizes, int n_in,
                              void* d_out, int out_size, void* d_ws, size_t ws_size,
                              hipStream_t stream) {
  (void)in_sizes; (void)n_in; (void)out_size; (void)ws_size;
  const float* x    = (const float*)d_in[0];
  const float* qw   = (const float*)d_in[1];
  const float* qb2  = (const float*)d_in[2];
  const float* q_s  = (const float*)d_in[3];
  const float* q_t  = (const float*)d_in[4];
  const float* kw   = (const float*)d_in[5];
  const float* kb2  = (const float*)d_in[6];
  const float* k_s  = (const float*)d_in[7];
  const float* k_t  = (const float*)d_in[8];
  const float* vw   = (const float*)d_in[9];
  const float* vb2  = (const float*)d_in[10];
  const float* v_s  = (const float*)d_in[11];
  const float* v_t  = (const float*)d_in[12];
  const float* lvw  = (const float*)d_in[13];
  const float* lvb  = (const float*)d_in[14];
  const float* lv_s = (const float*)d_in[15];
  const float* lv_t = (const float*)d_in[16];
  const float* th1w = (const float*)d_in[17];
  const float* th1b = (const float*)d_in[18];
  const float* th2w = (const float*)d_in[19];
  const float* th2b = (const float*)d_in[20];
  const float* ab   = (const float*)d_in[21];
  const float* pw   = (const float*)d_in[22];
  const float* pb   = (const float*)d_in[23];
  const float* p_s  = (const float*)d_in[24];
  const float* p_t  = (const float*)d_in[25];

  char* ws = (char*)d_ws;
  bf16_t* wqkv = (bf16_t*)(ws + 0);
  float*  bqkv = (float*)(ws + 1179648);
  bf16_t* pwb  = (bf16_t*)(ws + 1185792);
  float*  pbf  = (float*)(ws + 1972224);
  float*  lvwf = (float*)(ws + 1973760);
  float*  lvbf = (float*)(ws + 2010624);
  bf16_t* qbuf = (bf16_t*)(ws + 2014720);     // (B, NH, 208, 32) bf16
  bf16_t* kbuf = (bf16_t*)(ws + 15646208);    // (B, NH, 208, 32) bf16
  bf16_t* vbuf = (bf16_t*)(ws + 29277696);    // (B, 1024, 224) bf16
  bf16_t* vlb  = (bf16_t*)(ws + 87997952);    // (B, 1024, 208) bf16
  bf16_t* obuf = (bf16_t*)(ws + 142523904);   // (B, 208, 1024) bf16 token-major
  bf16_t* xbf  = (bf16_t*)(ws + 197049856);   // (B, 208, 384)  bf16 token-major

  prep_kernel<<<3888, 256, 0, stream>>>(qw, qb2, q_s, q_t, kw, kb2, k_s, k_t,
      vw, vb2, v_s, v_t, lvw, lvb, lv_s, lv_t, pw, pb, p_s, p_t,
      wqkv, bqkv, pwb, pbf, lvwf, lvbf);
  xpose_kernel<<<dim3(B_, 12), 256, 0, stream>>>(x, xbf);
  qkv_gemm<<<dim3(B_, 12), 256, 0, stream>>>(xbf, wqkv, bqkv, qbuf, kbuf, vbuf);
  dwconv_kernel<<<106496, 256, 0, stream>>>(vbuf, lvwf, lvbf, vlb);
  attn_kernel<<<dim3(B_, 13), 256, 172032, stream>>>(qbuf, kbuf, vbuf, vlb,
      ab, th1w, th1b, th2w, th2b, obuf);
  proj_kernel<<<dim3(B_, 3), 256, 0, stream>>>(obuf, pwb, pbf, (float*)d_out);
}
